// MoeBlock_16320875724817
// MI455X (gfx1250) — matrix operand order repaired and checked by reading, not compiled, not run on hardware
//
#include <hip/hip_runtime.h>
#include <hip/hip_bf16.h>

// ---------------------------------------------------------------------------
// MoE block (top-2, 64 experts) for gfx1250: bf16 WMMA grouped GEMMs,
// async-to-LDS A-tile staging, v_perm packed bf16 conversion, LDS ping-pong.
// B=4, S=2048 -> T=8192 tokens, EMB=1024, MLP=2048, rows = T*2 = 16384.
// ---------------------------------------------------------------------------

#define NUM_EXPERTS 64
#define T_TOKENS    8192
#define EMB         1024
#define MLP         2048
#define ROWS        (T_TOKENS * 2)
#define TILE_M      32
#define MAX_TILES   (ROWS / TILE_M + NUM_EXPERTS)   // 576

typedef __bf16 v16bf __attribute__((ext_vector_type(16)));
typedef float  v8f   __attribute__((ext_vector_type(8)));

// ---- workspace layout (bytes) ----
#define WS_COUNTS   0                         // 64 u32
#define WS_CURSORS  1024                      // 64 u32
#define WS_OFFSETS  2048                      // 65 u32
#define WS_TOTALS   4096                      // 1 u32
#define WS_DESC     8192                      // 576 int4
#define WS_SEL      32768                     // 16384 int
#define WS_WTS      131072                    // 16384 float
#define WS_ROWTOK   196608                    // 16384 int
#define WS_ROWWT    262144                    // 16384 float
#define WS_XBF      524288                    // 8192*1024 bf16 = 16 MiB
#define WS_INTER    (524288 + 16777216)       // 16384*2048 bf16 = 64 MiB

__device__ __forceinline__ unsigned short bf16u(float f) {
    unsigned u = __float_as_uint(f);
    return (unsigned short)((u + 0x7FFFu + ((u >> 16) & 1u)) >> 16);
}

// pack two floats into one dword of bf16 (low = a, high = b):
// round-half-up then a single v_perm_b32 byte-merge of the two high halves.
__device__ __forceinline__ unsigned pack_bf16x2(float a, float b) {
    unsigned ua = __float_as_uint(a) + 0x8000u;
    unsigned ub = __float_as_uint(b) + 0x8000u;
    return __builtin_amdgcn_perm(ub, ua, 0x07060302u);
}

__device__ __forceinline__ void wait_async0() {
#if __has_builtin(__builtin_amdgcn_s_wait_asynccnt)
    __builtin_amdgcn_s_wait_asynccnt(0);
#else
    asm volatile("s_wait_asynccnt 0x0" ::: "memory");
#endif
}

// async 16B copy: per-lane global address -> per-lane LDS address (ASYNCcnt)
__device__ __forceinline__ void async_copy_b128(unsigned lds_addr, unsigned long long gaddr) {
    asm volatile("global_load_async_to_lds_b128 %0, %1, off"
                 :: "v"(lds_addr), "v"(gaddr) : "memory");
}

// ---------------------------------------------------------------------------
// 1. fp32 -> bf16 conversion of the token activations
// ---------------------------------------------------------------------------
__global__ void __launch_bounds__(256) convert_x_kernel(
    const float* __restrict__ x, unsigned short* __restrict__ xb) {
    size_t i = ((size_t)blockIdx.x * 256 + threadIdx.x) * 4;
    float4 v = *(const float4*)(x + i);
    uint2 r;
    r.x = pack_bf16x2(v.x, v.y);
    r.y = pack_bf16x2(v.z, v.w);
    *(uint2*)(xb + i) = r;
}

// ---------------------------------------------------------------------------
// 2. Router: gate logits + top-2 + softmax, expert histogram
// ---------------------------------------------------------------------------
__global__ void __launch_bounds__(256) gate_topk_kernel(
    const float* __restrict__ x, const float* __restrict__ gk,
    int* __restrict__ sel, float* __restrict__ wts, unsigned* __restrict__ counts) {
    __shared__ float xs[EMB];
    __shared__ float part[256];
    __shared__ float logit[NUM_EXPERTS];
    int t = blockIdx.x;
    const float* xr = x + (size_t)t * EMB;
    {
        float4 v = ((const float4*)xr)[threadIdx.x];
        ((float4*)xs)[threadIdx.x] = v;
    }
    __syncthreads();
    int e = threadIdx.x & 63, ch = threadIdx.x >> 6;
    float s = 0.f;
    int k0 = ch * 256;
    for (int k = k0; k < k0 + 256; ++k)
        s = fmaf(xs[k], gk[(size_t)k * NUM_EXPERTS + e], s);
    part[threadIdx.x] = s;
    __syncthreads();
    if (threadIdx.x < 64)
        logit[e] = part[e] + part[e + 64] + part[e + 128] + part[e + 192];
    __syncthreads();
    if (threadIdx.x == 0) {
        int i1 = 0; float v1 = logit[0];
        for (int i = 1; i < 64; ++i) if (logit[i] > v1) { v1 = logit[i]; i1 = i; }
        int i2 = (i1 == 0) ? 1 : 0; float v2 = logit[i2];
        for (int i = 0; i < 64; ++i)
            if (i != i1 && logit[i] > v2) { v2 = logit[i]; i2 = i; }
        float ed = __expf(v2 - v1);
        float inv = 1.f / (1.f + ed);
        sel[t * 2] = i1; sel[t * 2 + 1] = i2;
        wts[t * 2] = inv; wts[t * 2 + 1] = ed * inv;
        atomicAdd(&counts[i1], 1u);
        atomicAdd(&counts[i2], 1u);
    }
}

// ---------------------------------------------------------------------------
// 3. Prefix sum + tile-descriptor build (single wave; trivial work)
// ---------------------------------------------------------------------------
__global__ void scan_build_kernel(
    const unsigned* __restrict__ counts, unsigned* __restrict__ offsets,
    int4* __restrict__ desc, unsigned* __restrict__ totals) {
    if (threadIdx.x == 0 && blockIdx.x == 0) {
        unsigned run = 0, tt = 0;
        for (int e = 0; e < NUM_EXPERTS; ++e) {
            offsets[e] = run;
            unsigned c = counts[e];
            unsigned nt = (c + TILE_M - 1) / TILE_M;
            for (unsigned i = 0; i < nt; ++i) {
                unsigned r0 = run + i * TILE_M;
                unsigned r1 = run + c;
                unsigned re = r0 + TILE_M < r1 ? r0 + TILE_M : r1;
                desc[tt++] = make_int4(e, (int)r0, (int)re, 0);
            }
            run += c;
        }
        offsets[NUM_EXPERTS] = run;
        totals[0] = tt;
    }
}

// ---------------------------------------------------------------------------
// 4. Scatter (token,k) pairs into expert-sorted row order
// ---------------------------------------------------------------------------
__global__ void __launch_bounds__(256) scatter_kernel(
    const int* __restrict__ sel, const float* __restrict__ wts,
    const unsigned* __restrict__ offsets, unsigned* __restrict__ cursors,
    int* __restrict__ row_token, float* __restrict__ row_weight) {
    int g = blockIdx.x * 256 + threadIdx.x;
    if (g >= ROWS) return;
    int e = sel[g];
    unsigned slot = offsets[e] + atomicAdd(&cursors[e], 1u);
    row_token[slot] = g >> 1;
    row_weight[slot] = wts[g];
}

// ---------------------------------------------------------------------------
// Fragment layout (CDNA5 bf16 WMMA operand layouts, wave32)
//   A 16x32: lane = m + 16*((k>>3)&1), dword j = ((k>>4)<<2)|((k&7)>>1), elem k&1
//            => a contiguous 16B global chunk (m, k-octet) is one 16B LDS chunk
//   B 32x16: lane = n + 16*((k>>3)&1), dword j = ((k>>4)<<2)|((k&7)>>1), elem k&1
//            => k-pair (2q,2q+1) for column n shares one LDS dword
//   C 16x16: vgpr r -> m = r + 8*(lane>>4), n = lane&15
// ---------------------------------------------------------------------------

// ---------------------------------------------------------------------------
// 5. FFN1: h = silu(x@w0) * (x@w1), grouped by expert; bf16 output to inter
//    grid = (MAX_TILES, MLP/128), block = 256 (8 waves), tile = 32Mx128N
//    LDS ping-pong: single barrier per K-step.
// ---------------------------------------------------------------------------
__global__ void __launch_bounds__(256) ffn1_kernel(
    const unsigned short* __restrict__ xbf,
    const float* __restrict__ w0, const float* __restrict__ w1,
    const int* __restrict__ row_token,
    const int4* __restrict__ desc, const unsigned* __restrict__ totals,
    unsigned short* __restrict__ inter) {
    __shared__ __align__(32) unsigned lds_a[2][2 * 32 * 8];        // 2 x 2 KiB
    __shared__ __align__(32) unsigned lds_b[2][2 * 8 * 32 * 8];    // 2 x 16 KiB
    unsigned tile = blockIdx.x;
    if (tile >= totals[0]) return;
    int4 d = desc[tile];
    int e = d.x, row0 = d.y, row_end = d.z;
    int nb = blockIdx.y;                        // 128-col block of MLP
    int lane = threadIdx.x & 31, wv = threadIdx.x >> 5;

    v8f acc00 = {0.f,0.f,0.f,0.f,0.f,0.f,0.f,0.f};
    v8f acc01 = acc00, acc10 = acc00, acc11 = acc00;

    const float* w0e = w0 + (size_t)e * EMB * MLP + (size_t)nb * 128;
    const float* w1e = w1 + (size_t)e * EMB * MLP + (size_t)nb * 128;

    // ---- wave-0 async A-staging setup: 128 x 16B chunks (32 rows x 4 octets)
    unsigned a_lds[4]; unsigned long long a_g[4];
    if (wv == 0) {
        #pragma unroll
        for (int i = 0; i < 4; ++i) {
            int c = lane + 32 * i;              // chunk 0..127
            int m = c >> 2, seg = c & 3;        // row, k-octet
            int row = row0 + m;
            int tok = row_token[row < row_end ? row : row0];
            a_g[i] = (unsigned long long)(uintptr_t)(xbf + (size_t)tok * EMB) +
                     (unsigned long long)seg * 16;
            int lane_t = (m & 15) + 16 * (seg & 1);
            int j0 = (seg >> 1) * 4;
            a_lds[i] = (unsigned)(uintptr_t)&lds_a[0][((m >> 4) * 32 + lane_t) * 8 + j0];
        }
    }
    // ---- B-staging role: thread = (mat, k-pair q, ntile) ----
    int matB  = threadIdx.x >> 7;               // 0..1
    int qB    = (threadIdx.x >> 3) & 15;        // k-pair: rows 2q, 2q+1
    int ntile = threadIdx.x & 7;
    int lnn16 = 16 * ((qB >> 2) & 1);           // lane bias from (k>>3)&1
    int jB    = ((qB >> 3) << 2) | (qB & 3);

    for (int kc = 0; kc < EMB / 32; ++kc) {
        int p = kc & 1;
        // ---- A tile via async global->LDS (swizzled), wave 0 only ----
        if (wv == 0) {
            unsigned aoff = p ? (unsigned)sizeof(lds_a[0]) : 0u;
            #pragma unroll
            for (int i = 0; i < 4; ++i)
                async_copy_b128(a_lds[i] + aoff, a_g[i] + (unsigned long long)kc * 64);
        }
        // ---- B0,B1 tiles (32K x 128N): fp32 -> packed bf16 dwords ----
        {
            const float* wbase = matB ? w1e : w0e;
            const float* s = wbase + (size_t)(kc * 32 + 2 * qB) * MLP + ntile * 16;
            float f0[16], f1[16];
            #pragma unroll
            for (int q = 0; q < 4; ++q) {
                float4 t0 = ((const float4*)s)[q];
                float4 t1 = ((const float4*)(s + MLP))[q];
                f0[4*q] = t0.x; f0[4*q+1] = t0.y; f0[4*q+2] = t0.z; f0[4*q+3] = t0.w;
                f1[4*q] = t1.x; f1[4*q+1] = t1.y; f1[4*q+2] = t1.z; f1[4*q+3] = t1.w;
            }
            #pragma unroll
            for (int i = 0; i < 16; ++i)
                lds_b[p][(((matB * 8 + ntile) * 32) + (i + lnn16)) * 8 + jB] =
                    pack_bf16x2(f0[i], f1[i]);
            if (kc + 1 < EMB / 32)
                __builtin_prefetch(s + (size_t)32 * MLP, 0, 1);
        }
        wait_async0();
        __syncthreads();
        v16bf a0 = *(const v16bf*)&lds_a[p][(0 * 32 + lane) * 8];
        v16bf a1 = *(const v16bf*)&lds_a[p][(1 * 32 + lane) * 8];
        v16bf b0 = *(const v16bf*)&lds_b[p][(((0 * 8 + wv) * 32) + lane) * 8];
        v16bf b1 = *(const v16bf*)&lds_b[p][(((1 * 8 + wv) * 32) + lane) * 8];
        acc00 = __builtin_amdgcn_wmma_f32_16x16x32_bf16(false, a0, false, b0, (short)0, acc00, false, false);
        acc01 = __builtin_amdgcn_wmma_f32_16x16x32_bf16(false, a0, false, b1, (short)0, acc01, false, false);
        acc10 = __builtin_amdgcn_wmma_f32_16x16x32_bf16(false, a1, false, b0, (short)0, acc10, false, false);
        acc11 = __builtin_amdgcn_wmma_f32_16x16x32_bf16(false, a1, false, b1, (short)0, acc11, false, false);
    }

    // epilogue: inter = silu(h0) * h1  (bf16)
    int ncol = nb * 128 + wv * 16 + (lane & 15);
    int mhalf = 8 * (lane >> 4);
    #pragma unroll
    for (int mt = 0; mt < 2; ++mt) {
        v8f g = mt ? acc10 : acc00;
        v8f u = mt ? acc11 : acc01;
        #pragma unroll
        for (int r = 0; r < 8; ++r) {
            int row = row0 + mt * 16 + mhalf + r;
            if (row < row_end) {
                float gv = g[r];
                float sv = (gv / (1.f + __expf(-gv))) * u[r];
                inter[(size_t)row * MLP + ncol] = bf16u(sv);
            }
        }
    }
}

// ---------------------------------------------------------------------------
// 6. FFN2: out_sorted = inter @ wo, scaled by routing weight, atomic combine
//    grid = (MAX_TILES, EMB/128), block = 256; LDS ping-pong, one barrier/step
// ---------------------------------------------------------------------------
__global__ void __launch_bounds__(256) ffn2_kernel(
    const unsigned short* __restrict__ inter,
    const float* __restrict__ wo,
    const int* __restrict__ row_token, const float* __restrict__ row_weight,
    const int4* __restrict__ desc, const unsigned* __restrict__ totals,
    float* __restrict__ out) {
    __shared__ __align__(32) unsigned lds_a[2][2 * 32 * 8];    // 2 x 2 KiB
    __shared__ __align__(32) unsigned lds_b[2][8 * 32 * 8];    // 2 x 8 KiB
    unsigned tile = blockIdx.x;
    if (tile >= totals[0]) return;
    int4 d = desc[tile];
    int e = d.x, row0 = d.y, row_end = d.z;
    int nb = blockIdx.y;
    int lane = threadIdx.x & 31, wv = threadIdx.x >> 5;

    v8f acc0 = {0.f,0.f,0.f,0.f,0.f,0.f,0.f,0.f};
    v8f acc1 = acc0;

    const float* woe = wo + (size_t)e * MLP * EMB + (size_t)nb * 128;

    unsigned a_lds[4]; unsigned long long a_g[4];
    if (wv == 0) {
        #pragma unroll
        for (int i = 0; i < 4; ++i) {
            int c = lane + 32 * i;
            int m = c >> 2, seg = c & 3;
            int row = row0 + m;
            int rr = row < row_end ? row : row0;
            a_g[i] = (unsigned long long)(uintptr_t)(inter + (size_t)rr * MLP) +
                     (unsigned long long)seg * 16;
            int lane_t = (m & 15) + 16 * (seg & 1);
            int j0 = (seg >> 1) * 4;
            a_lds[i] = (unsigned)(uintptr_t)&lds_a[0][((m >> 4) * 32 + lane_t) * 8 + j0];
        }
    }
    int qB    = (threadIdx.x >> 3) & 15;
    int ntile = threadIdx.x & 7;
    int lnn16 = 16 * ((qB >> 2) & 1);
    int jB    = ((qB >> 3) << 2) | (qB & 3);
    bool doB  = threadIdx.x < 128;

    for (int kc = 0; kc < MLP / 32; ++kc) {
        int p = kc & 1;
        if (wv == 0) {
            unsigned aoff = p ? (unsigned)sizeof(lds_a[0]) : 0u;
            #pragma unroll
            for (int i = 0; i < 4; ++i)
                async_copy_b128(a_lds[i] + aoff, a_g[i] + (unsigned long long)kc * 64);
        }
        if (doB) {
            const float* s = woe + (size_t)(kc * 32 + 2 * qB) * EMB + ntile * 16;
            float f0[16], f1[16];
            #pragma unroll
            for (int q = 0; q < 4; ++q) {
                float4 t0 = ((const float4*)s)[q];
                float4 t1 = ((const float4*)(s + EMB))[q];
                f0[4*q] = t0.x; f0[4*q+1] = t0.y; f0[4*q+2] = t0.z; f0[4*q+3] = t0.w;
                f1[4*q] = t1.x; f1[4*q+1] = t1.y; f1[4*q+2] = t1.z; f1[4*q+3] = t1.w;
            }
            #pragma unroll
            for (int i = 0; i < 16; ++i)
                lds_b[p][((ntile * 32) + (i + lnn16)) * 8 + jB] = pack_bf16x2(f0[i], f1[i]);
            if (kc + 1 < MLP / 32)
                __builtin_prefetch(s + (size_t)32 * EMB, 0, 1);
        }
        wait_async0();
        __syncthreads();
        v16bf a0 = *(const v16bf*)&lds_a[p][(0 * 32 + lane) * 8];
        v16bf a1 = *(const v16bf*)&lds_a[p][(1 * 32 + lane) * 8];
        v16bf b  = *(const v16bf*)&lds_b[p][((wv * 32) + lane) * 8];
        acc0 = __builtin_amdgcn_wmma_f32_16x16x32_bf16(false, a0, false, b, (short)0, acc0, false, false);
        acc1 = __builtin_amdgcn_wmma_f32_16x16x32_bf16(false, a1, false, b, (short)0, acc1, false, false);
    }

    // epilogue: weighted atomic combine into out[token]
    int ncol = nb * 128 + wv * 16 + (lane & 15);
    int mhalf = 8 * (lane >> 4);
    #pragma unroll
    for (int mt = 0; mt < 2; ++mt) {
        v8f a = mt ? acc1 : acc0;
        #pragma unroll
        for (int r = 0; r < 8; ++r) {
            int row = row0 + mt * 16 + mhalf + r;
            if (row < row_end) {
                float v = a[r] * row_weight[row];
                atomicAdd(&out[(size_t)row_token[row] * EMB + ncol], v);
            }
        }
    }
}

// ---------------------------------------------------------------------------
extern "C" void kernel_launch(void* const* d_in, const int* in_sizes, int n_in,
                              void* d_out, int out_size, void* d_ws, size_t ws_size,
                              hipStream_t stream) {
    const float* x  = (const float*)d_in[0];   // [4,2048,1024]
    const float* gk = (const float*)d_in[1];   // [1024,64]
    const float* w0 = (const float*)d_in[2];   // [64,1024,2048]
    const float* w1 = (const float*)d_in[3];   // [64,1024,2048]
    const float* wo = (const float*)d_in[4];   // [64,2048,1024]
    float* out = (float*)d_out;                // [4,2048,1024] fp32

    char* ws = (char*)d_ws;
    unsigned*       counts   = (unsigned*)(ws + WS_COUNTS);
    unsigned*       cursors  = (unsigned*)(ws + WS_CURSORS);
    unsigned*       offsets  = (unsigned*)(ws + WS_OFFSETS);
    unsigned*       totals   = (unsigned*)(ws + WS_TOTALS);
    int4*           desc     = (int4*)(ws + WS_DESC);
    int*            sel      = (int*)(ws + WS_SEL);
    float*          wts      = (float*)(ws + WS_WTS);
    int*            row_tok  = (int*)(ws + WS_ROWTOK);
    float*          row_wt   = (float*)(ws + WS_ROWWT);
    unsigned short* xbf      = (unsigned short*)(ws + WS_XBF);
    unsigned short* inter    = (unsigned short*)(ws + WS_INTER);

    hipMemsetAsync(ws, 0, 8192, stream);
    hipMemsetAsync(out, 0, (size_t)T_TOKENS * EMB * sizeof(float), stream);

    convert_x_kernel<<<(T_TOKENS * EMB) / (256 * 4), 256, 0, stream>>>(x, xbf);
    gate_topk_kernel<<<T_TOKENS, 256, 0, stream>>>(x, gk, sel, wts, counts);
    scan_build_kernel<<<1, 64, 0, stream>>>(counts, offsets, desc, totals);
    scatter_kernel<<<ROWS / 256, 256, 0, stream>>>(sel, wts, offsets, cursors, row_tok, row_wt);
    ffn1_kernel<<<dim3(MAX_TILES, MLP / 128), 256, 0, stream>>>(
        xbf, w0, w1, row_tok, desc, totals, inter);
    ffn2_kernel<<<dim3(MAX_TILES, EMB / 128), 256, 0, stream>>>(
        inter, wo, row_tok, row_wt, desc, totals, out);
}
